// ODELSTM_35072702939740
// MI455X (gfx1250) — compile-verified, matrix operand order checked by reading, not run on hardware
//
#include <hip/hip_runtime.h>
#include <hip/hip_bf16.h>

// ---------------------------------------------------------------------------
// ODE-LSTM persistent kernel for gfx1250 (MI455X).
// B=512, S=256, I=128, H=OH=768.
// One workgroup (16 waves / 512 threads) owns 16 batch rows for all 256 steps.
// State (h, c, ksum) in LDS f32; WMMA operands in LDS bf16.
// Gate GEMM: fused K=896, split into two 2-gate even/odd pipelines to stay
// under the VGPR budget (no scratch spills). RK4 GEMMs: 3-set pipelines.
// ---------------------------------------------------------------------------

#define Bdim 512
#define Sdim 256
#define Idim 128
#define Hdim 768
#define KC   (Idim + Hdim)        // 896: fused gate K
#define ROWS 16                   // batch rows per workgroup
#define NWAVES 16
#define NTILES (Hdim / 16)        // 48
#define TPW (NTILES / NWAVES)     // 3 N-tiles per wave per GEMM phase
#define GSLAB (KC / 32)           // 28 K-slabs in gate GEMM
#define RSLAB (Hdim / 32)         // 24 K-slabs in RK4 GEMMs

typedef __attribute__((ext_vector_type(16))) __bf16 v16bf;
typedef __attribute__((ext_vector_type(8)))  __bf16 v8bf;
typedef __attribute__((ext_vector_type(8)))  float  v8f;

static __device__ __forceinline__ __bf16 bf16_rn(float f) {
    unsigned int u = __builtin_bit_cast(unsigned int, f);
    unsigned int r = u + 0x7FFFu + ((u >> 16) & 1u);   // round-to-nearest-even
    unsigned short s = (unsigned short)(r >> 16);
    return __builtin_bit_cast(__bf16, s);
}

// rcp-based activations: avoids v_div_fmas/div_fixup sequences.
static __device__ __forceinline__ float fast_sigmoid(float x) {
    float xc = fminf(fmaxf(x, -30.0f), 30.0f);
    return __builtin_amdgcn_rcpf(1.0f + __expf(-xc));
}
static __device__ __forceinline__ float fast_tanh(float x) {
    float xc = fminf(fmaxf(x, -15.0f), 15.0f);
    float e  = __expf(2.0f * xc);
    return (e - 1.0f) * __builtin_amdgcn_rcpf(e + 1.0f);
}

// 16x32 bf16 A-fragment (or 32x16 B-fragment; same per-lane mapping):
// lane%16 = row (A) / col (B); lane/16 = K half.
static __device__ __forceinline__ v16bf ldfrag(const __bf16* base, int ld,
                                               int lrow, int sel) {
    const __bf16* q = base + (size_t)lrow * ld + sel * 8;
    v8bf lo = *(const v8bf*)(q);
    v8bf hi = *(const v8bf*)(q + 16);
    return __builtin_shufflevector(lo, hi, 0, 1, 2, 3, 4, 5, 6, 7,
                                            8, 9, 10, 11, 12, 13, 14, 15);
}

static __device__ __forceinline__ v8f wmma_bf16(v16bf a, v16bf b, v8f c) {
    return __builtin_amdgcn_wmma_f32_16x16x32_bf16(
        false, a, false, b, (short)0, c, false, false);
}

// ---------------------------------------------------------------------------
// Prologue kernels: fp32 -> bf16 weight staging in workspace.
// ---------------------------------------------------------------------------
__global__ void cvt_f32_bf16(const float* __restrict__ src,
                             __bf16* __restrict__ dst, int n) {
    int i = blockIdx.x * blockDim.x + threadIdx.x;
    int stride = gridDim.x * blockDim.x;
    for (; i < n; i += stride) dst[i] = bf16_rn(src[i]);
}

// Wcat[n, 0:128] = W_ih[n, :] ; Wcat[n, 128:896] = W_hh[n, :]   (4H x 896)
__global__ void build_wcat(const float* __restrict__ wih,
                           const float* __restrict__ whh,
                           __bf16* __restrict__ dst) {
    const int n_tot = 4 * Hdim * KC;
    int i = blockIdx.x * blockDim.x + threadIdx.x;
    int stride = gridDim.x * blockDim.x;
    for (; i < n_tot; i += stride) {
        int n = i / KC, k = i - n * KC;
        float v = (k < Idim) ? wih[(size_t)n * Idim + k]
                             : whh[(size_t)n * Hdim + (k - Idim)];
        dst[i] = bf16_rn(v);
    }
}

// ---------------------------------------------------------------------------
// Main persistent kernel.
// ---------------------------------------------------------------------------
__global__ __launch_bounds__(NWAVES * 32, 1)
void odelstm_persistent(const float* __restrict__ inputs,     // (B,S,I)
                        const float* __restrict__ timespans,  // (B,S): row 0
                        const float* __restrict__ b_ih,       // (4H)
                        const float* __restrict__ b_hh,       // (4H)
                        const float* __restrict__ b1v,        // (OH)
                        const float* __restrict__ b2v,        // (H)
                        const __bf16* __restrict__ Wcat,      // (4H, 896) bf16
                        const __bf16* __restrict__ W1,        // (OH, H) bf16
                        const __bf16* __restrict__ W2,        // (H, OH) bf16
                        float* __restrict__ out)              // outputs|h|c
{
    extern __shared__ char smem[];
    float*  hF   = (float*)smem;                   // 16x768 f32 (current h)
    float*  cF   = hF + ROWS * Hdim;               // 16x768 f32 (current c)
    float*  ksum = cF + ROWS * Hdim;               // 16x768 f32 (RK4 sum)
    __bf16* Acat = (__bf16*)(ksum + ROWS * Hdim);  // 16x896 bf16: [x | h/z]
    __bf16* zb   = Acat + ROWS * KC;               // 16x768 bf16 (y buffer)

    const int tid  = threadIdx.x;
    const int lane = tid & 31;
    const int wave = tid >> 5;
    const int lrow = lane & 15;   // row (A) / col (B) / N-col (C/D)
    const int sel  = lane >> 4;   // K half (A/B); M half (C/D)
    const int b0   = blockIdx.x * ROWS;

    // ---- loop-invariant biases (per-lane, per owned tile; const indices) ----
    float gbias[TPW][4], bz1[TPW], bz2[TPW];
#pragma unroll
    for (int j = 0; j < TPW; ++j) {
        const int ncol = (wave + j * NWAVES) * 16 + lrow;
#pragma unroll
        for (int g = 0; g < 4; ++g)
            gbias[j][g] = b_ih[g * Hdim + ncol] + b_hh[g * Hdim + ncol];
        bz1[j] = b1v[ncol];
        bz2[j] = b2v[ncol];
    }

    // ---- init: h = c = 0; load x_0 tile ----
    const __bf16 bzero = bf16_rn(0.0f);
    for (int i = tid; i < ROWS * Hdim; i += blockDim.x) {
        hF[i] = 0.0f; cF[i] = 0.0f;
        const int m = i / Hdim, n = i - m * Hdim;
        Acat[m * KC + Idim + n] = bzero;
    }
    for (int i = tid; i < ROWS * Idim; i += blockDim.x) {
        const int m = i / Idim, k = i - m * Idim;
        Acat[m * KC + k] =
            bf16_rn(__builtin_nontemporal_load(
                &inputs[((size_t)(b0 + m) * Sdim + 0) * Idim + k]));
    }
    __syncthreads();

    for (int t = 0; t < Sdim; ++t) {
        const float dt = timespans[t];  // timespans[0, t]

        // ====== LSTM gates: fused K=896, two 2-gate even/odd pipelines ======
#pragma unroll
        for (int j = 0; j < TPW; ++j) {
            const int ntile = wave + j * NWAVES;
            const int ncol  = ntile * 16 + lrow;
            const __bf16* Wb = Wcat + (size_t)(ntile * 16) * KC;
            v8f acc[4];

#pragma unroll
            for (int p = 0; p < 2; ++p) {           // pass 0: i,f  pass 1: g,o
                const __bf16* W0 = Wb + (size_t)(2 * p) * Hdim * KC;
                const __bf16* W1g = Wb + (size_t)(2 * p + 1) * Hdim * KC;
                v8f ac0 = v8f(0.f), ac1 = v8f(0.f);
                v16bf aA, aB, b0A, b1A, b0B, b1B;

                aA  = ldfrag(Acat, KC, lrow, sel);
                b0A = ldfrag(W0,  KC, lrow, sel);
                b1A = ldfrag(W1g, KC, lrow, sel);
                for (int s = 0; s < GSLAB - 2; s += 2) {
                    const int kB = (s + 1) * 32, kA = (s + 2) * 32;
                    aB  = ldfrag(Acat + kB, KC, lrow, sel);
                    b0B = ldfrag(W0 + kB,  KC, lrow, sel);
                    b1B = ldfrag(W1g + kB, KC, lrow, sel);
                    ac0 = wmma_bf16(aA, b0A, ac0);
                    ac1 = wmma_bf16(aA, b1A, ac1);
                    aA  = ldfrag(Acat + kA, KC, lrow, sel);
                    b0A = ldfrag(W0 + kA,  KC, lrow, sel);
                    b1A = ldfrag(W1g + kA, KC, lrow, sel);
                    ac0 = wmma_bf16(aB, b0B, ac0);
                    ac1 = wmma_bf16(aB, b1B, ac1);
                }
                {   // epilogue: set A holds slab GSLAB-2; finish last slab
                    const int kB = (GSLAB - 1) * 32;
                    aB  = ldfrag(Acat + kB, KC, lrow, sel);
                    b0B = ldfrag(W0 + kB,  KC, lrow, sel);
                    b1B = ldfrag(W1g + kB, KC, lrow, sel);
                    ac0 = wmma_bf16(aA, b0A, ac0);
                    ac1 = wmma_bf16(aA, b1A, ac1);
                    ac0 = wmma_bf16(aB, b0B, ac0);
                    ac1 = wmma_bf16(aB, b1B, ac1);
                }
                acc[2 * p]     = ac0;
                acc[2 * p + 1] = ac1;
            }

            // cell update; bf16(h2) -> zb (unused during gate phase)
#pragma unroll
            for (int r = 0; r < 8; ++r) {
                const int m = r + sel * 8;
                const float gi = fast_sigmoid(acc[0][r] + gbias[j][0]);
                const float gf = fast_sigmoid(acc[1][r] + gbias[j][1]);
                const float gg = fast_tanh   (acc[2][r] + gbias[j][2]);
                const float go = fast_sigmoid(acc[3][r] + gbias[j][3]);
                const float c2 = gf * cF[m * Hdim + ncol] + gi * gg;
                const float h2 = go * fast_tanh(c2);
                cF[m * Hdim + ncol] = c2;
                hF[m * Hdim + ncol] = h2;
                zb[m * Hdim + ncol] = bf16_rn(h2);
            }
        }
        __syncthreads();

        // ====== RK4: y in zb (ld=H), z in Acat h-region (ld=KC) ======
        __bf16* zreg = Acat + Idim;
        for (int e = 0; e < 4; ++e) {
            const float wv = (e == 1 || e == 2) ? 2.0f : 1.0f;  // RK weights
            const float av = (e == 2) ? 1.0f : 0.5f;            // next-y coeff

            // GEMM1: z = tanh(y @ W1^T + b1); 3-set distance-3 pipeline
#pragma unroll
            for (int j = 0; j < TPW; ++j) {
                const int ntile = wave + j * NWAVES;
                const int ncol  = ntile * 16 + lrow;
                const __bf16* Wb = W1 + (size_t)(ntile * 16) * Hdim;
                v8f acc = v8f(0.f);
                v16bf a0, a1, a2, w0, w1, w2;
                a0 = ldfrag(zb, Hdim, lrow, sel);      w0 = ldfrag(Wb, Hdim, lrow, sel);
                a1 = ldfrag(zb + 32, Hdim, lrow, sel); w1 = ldfrag(Wb + 32, Hdim, lrow, sel);
                for (int s = 0; s <= RSLAB - 6; s += 3) {
                    a2 = ldfrag(zb + (s + 2) * 32, Hdim, lrow, sel);
                    w2 = ldfrag(Wb + (s + 2) * 32, Hdim, lrow, sel);
                    acc = wmma_bf16(a0, w0, acc);
                    a0 = ldfrag(zb + (s + 3) * 32, Hdim, lrow, sel);
                    w0 = ldfrag(Wb + (s + 3) * 32, Hdim, lrow, sel);
                    acc = wmma_bf16(a1, w1, acc);
                    a1 = ldfrag(zb + (s + 4) * 32, Hdim, lrow, sel);
                    w1 = ldfrag(Wb + (s + 4) * 32, Hdim, lrow, sel);
                    acc = wmma_bf16(a2, w2, acc);
                }
                a2 = ldfrag(zb + (RSLAB - 1) * 32, Hdim, lrow, sel);
                w2 = ldfrag(Wb + (RSLAB - 1) * 32, Hdim, lrow, sel);
                acc = wmma_bf16(a0, w0, acc);
                acc = wmma_bf16(a1, w1, acc);
                acc = wmma_bf16(a2, w2, acc);
#pragma unroll
                for (int r = 0; r < 8; ++r) {
                    const int m = r + sel * 8;
                    zreg[m * KC + ncol] = bf16_rn(fast_tanh(acc[r] + bz1[j]));
                }
            }
            __syncthreads();

            // GEMM2: k = z @ W2^T + b2; fold into ksum, y_next -> zb
#pragma unroll
            for (int j = 0; j < TPW; ++j) {
                const int ntile = wave + j * NWAVES;
                const int ncol  = ntile * 16 + lrow;
                const __bf16* Wb = W2 + (size_t)(ntile * 16) * Hdim;
                v8f acc = v8f(0.f);
                v16bf a0, a1, a2, w0, w1, w2;
                a0 = ldfrag(zreg, KC, lrow, sel);      w0 = ldfrag(Wb, Hdim, lrow, sel);
                a1 = ldfrag(zreg + 32, KC, lrow, sel); w1 = ldfrag(Wb + 32, Hdim, lrow, sel);
                for (int s = 0; s <= RSLAB - 6; s += 3) {
                    a2 = ldfrag(zreg + (s + 2) * 32, KC, lrow, sel);
                    w2 = ldfrag(Wb + (s + 2) * 32, Hdim, lrow, sel);
                    acc = wmma_bf16(a0, w0, acc);
                    a0 = ldfrag(zreg + (s + 3) * 32, KC, lrow, sel);
                    w0 = ldfrag(Wb + (s + 3) * 32, Hdim, lrow, sel);
                    acc = wmma_bf16(a1, w1, acc);
                    a1 = ldfrag(zreg + (s + 4) * 32, KC, lrow, sel);
                    w1 = ldfrag(Wb + (s + 4) * 32, Hdim, lrow, sel);
                    acc = wmma_bf16(a2, w2, acc);
                }
                a2 = ldfrag(zreg + (RSLAB - 1) * 32, KC, lrow, sel);
                w2 = ldfrag(Wb + (RSLAB - 1) * 32, Hdim, lrow, sel);
                acc = wmma_bf16(a0, w0, acc);
                acc = wmma_bf16(a1, w1, acc);
                acc = wmma_bf16(a2, w2, acc);
#pragma unroll
                for (int r = 0; r < 8; ++r) {
                    const int m   = r + sel * 8;
                    const int idx = m * Hdim + ncol;
                    const float kv = acc[r] + bz2[j];
                    ksum[idx] = (e == 0) ? kv : ksum[idx] + wv * kv;
                    if (e < 3)
                        zb[idx] = bf16_rn(hF[idx] + av * dt * kv);
                }
            }
            __syncthreads();
        }

        // ==== h += dt/6*ksum; emit output; stage bf16(h) + x_{t+1} ====
        const float s6 = dt * (1.0f / 6.0f);
        for (int i = tid; i < ROWS * Hdim; i += blockDim.x) {
            const int m = i / Hdim, n = i - m * Hdim;
            const float hn = hF[i] + s6 * ksum[i];
            hF[i] = hn;
            Acat[m * KC + Idim + n] = bf16_rn(hn);  // gate A operand
            __builtin_nontemporal_store(
                hn, &out[((size_t)(b0 + m) * Sdim + t) * Hdim + n]);
        }
        if (t + 1 < Sdim) {
            for (int i = tid; i < ROWS * Idim; i += blockDim.x) {
                const int m = i / Idim, k = i - m * Idim;
                Acat[m * KC + k] =
                    bf16_rn(__builtin_nontemporal_load(
                        &inputs[((size_t)(b0 + m) * Sdim + (t + 1)) * Idim + k]));
            }
        }
        __syncthreads();
    }

    // ---- final h, c ----
    float* outH = out + (size_t)Bdim * Sdim * Hdim;
    float* outC = outH + (size_t)Bdim * Hdim;
    for (int i = tid; i < ROWS * Hdim; i += blockDim.x) {
        const int m = i / Hdim, n = i - m * Hdim;
        outH[(size_t)(b0 + m) * Hdim + n] = hF[i];
        outC[(size_t)(b0 + m) * Hdim + n] = cF[i];
    }
}

// ---------------------------------------------------------------------------
// Host launch
// ---------------------------------------------------------------------------
extern "C" void kernel_launch(void* const* d_in, const int* in_sizes, int n_in,
                              void* d_out, int out_size, void* d_ws, size_t ws_size,
                              hipStream_t stream) {
    (void)in_sizes; (void)n_in; (void)out_size; (void)ws_size;
    const float* inputs    = (const float*)d_in[0];
    const float* timespans = (const float*)d_in[1];
    const float* W_ih      = (const float*)d_in[2];  // (3072, 128)
    const float* W_hh      = (const float*)d_in[3];  // (3072, 768)
    const float* b_ih      = (const float*)d_in[4];
    const float* b_hh      = (const float*)d_in[5];
    const float* W1        = (const float*)d_in[6];  // (768, 768)
    const float* b1        = (const float*)d_in[7];
    const float* W2        = (const float*)d_in[8];  // (768, 768)
    const float* b2        = (const float*)d_in[9];

    const int nWcat = 4 * Hdim * KC;    // 2752512
    const int nW1   = Hdim * Hdim;      // 589824
    const int nW2   = Hdim * Hdim;      // 589824

    __bf16* WcatB = (__bf16*)d_ws;
    __bf16* W1B   = WcatB + nWcat;
    __bf16* W2B   = W1B + nW1;

    build_wcat<<<(nWcat + 255) / 256, 256, 0, stream>>>(W_ih, W_hh, WcatB);
    cvt_f32_bf16<<<(nW1 + 255) / 256, 256, 0, stream>>>(W1, W1B, nW1);
    cvt_f32_bf16<<<(nW2 + 255) / 256, 256, 0, stream>>>(W2, W2B, nW2);

    const size_t lds_bytes =
        3u * ROWS * Hdim * sizeof(float) +    // hF, cF, ksum
        (size_t)ROWS * KC * sizeof(__bf16) +  // Acat
        (size_t)ROWS * Hdim * sizeof(__bf16); // zb    => 200704 B

    static bool attr_set = false;
    if (!attr_set) {
        hipFuncSetAttribute((const void*)odelstm_persistent,
                            hipFuncAttributeMaxDynamicSharedMemorySize,
                            (int)lds_bytes);
        attr_set = true;
    }

    odelstm_persistent<<<dim3(Bdim / ROWS), dim3(NWAVES * 32), lds_bytes, stream>>>(
        inputs, timespans, b_ih, b_hh, b1, b2,
        WcatB, W1B, W2B, (float*)d_out);
}
